// PixelPropagationModule_14534169329983
// MI455X (gfx1250) — compile-verified, matrix operand order checked by reading, not run on hardware
//
#include <hip/hip_runtime.h>

// ---------------------------------------------------------------------------
// PixelPropagationModule on MI455X (gfx1250), wave32 + bf16 WMMA.
//   b=16, c=256, h=w=48, hw=2304.
//   Xa = flat view (hw x c), Xb = flat view (c x hw)  (same buffer!)
//   cos = Xa@Xb / n^2 ; S = relu(cos)^2 = relu(raw)^2 / n^4
//   TR  = (W@X + bias) stored row-major (c x hw) == raw-reshape flat view
//   Y   = S @ TRview(hw x c), written flat (== reshape to (c,h,w))
// Fusion: never materialize S (340MB). Flash-style: cos tile -> relu^2 -> Y.
// ---------------------------------------------------------------------------

#define CC 256
#define HW 2304
#define PB (CC * HW)   // 589824 elements per batch

typedef __bf16 bf16x16 __attribute__((ext_vector_type(16)));
typedef __bf16 bf16x2  __attribute__((ext_vector_type(2)));
typedef float  v8f     __attribute__((ext_vector_type(8)));
typedef float  f32x2   __attribute__((ext_vector_type(2)));

union Frag {
    unsigned int u[8];
    bf16x16      v;
};

// Packed f32x2 -> bf16x2 (hardware v_cvt_pk_bf16_f32, RNE).
__device__ __forceinline__ unsigned int cvt_pk2(float a, float b) {
    f32x2 f; f[0] = a; f[1] = b;
    bf16x2 h = __builtin_convertvector(f, bf16x2);
    return __builtin_bit_cast(unsigned int, h);
}

__device__ __forceinline__ unsigned short cvt1(float a) {
    __bf16 h = (__bf16)a;
    return __builtin_bit_cast(unsigned short, h);
}

// Documented 16-bit 16x32 A-matrix layout: element pair v (0..7) of a lane
// holds K = 16*(v>>2) + 8*half + 2*(v&3) + {0,1}.  (half = lane>=16)
__device__ __forceinline__ int kpat(int v, int half) {
    return ((v >> 2) << 4) + (half << 3) + ((v & 3) << 1);
}

// Fragment read from LDS (row base bp, u16 units): elements [8*half..8*half+7]
// and [16+8*half..+7] are contiguous -> two 16B-aligned ds_load_b128.
__device__ __forceinline__ void load_frag(const unsigned short* bp, int half, Frag& f) {
    uint4 a = *(const uint4*)(bp + 8 * half);
    uint4 b = *(const uint4*)(bp + 16 + 8 * half);
    f.u[0] = a.x; f.u[1] = a.y; f.u[2] = a.z; f.u[3] = a.w;
    f.u[4] = b.x; f.u[5] = b.y; f.u[6] = b.z; f.u[7] = b.w;
}

__device__ __forceinline__ v8f wmma_bf16(bf16x16 a, bf16x16 b, v8f c) {
    // (neg_a, A, neg_b, B, c_mod, C, reuse_a, reuse_b)
    return __builtin_amdgcn_wmma_f32_16x16x32_bf16(false, a, false, b,
                                                   (short)0, c, false, false);
}

// ---------------------------------------------------------------------------
// Kernel 1: per-batch 1/||x||^4
// ---------------------------------------------------------------------------
__global__ __launch_bounds__(256) void sumsq_kernel(const float* __restrict__ x,
                                                    float* __restrict__ inv_n4) {
    const float4* p = (const float4*)(x + (size_t)blockIdx.x * PB);
    float acc = 0.f;
    for (int i = threadIdx.x; i < PB / 4; i += 256) {
        float4 v = p[i];
        acc += v.x * v.x + v.y * v.y + v.z * v.z + v.w * v.w;
    }
    __shared__ float red[256];
    red[threadIdx.x] = acc;
    __syncthreads();
    for (int s = 128; s > 0; s >>= 1) {
        if (threadIdx.x < s) red[threadIdx.x] += red[threadIdx.x + s];
        __syncthreads();
    }
    if (threadIdx.x == 0) {
        float n2 = red[0];
        inv_n4[blockIdx.x] = 1.0f / (n2 * n2);
    }
}

// ---------------------------------------------------------------------------
// Kernel 2: TR(b, o, p) = sum_k W[o,k]*x[b,k,p] + bias[o]   (256 x 2304 GEMM)
// Block = 4 waves. Block tile: 64(o) x 64(p). Wave: 16(o) x 64(p) = 4 C tiles.
// A (W rows, bf16) kept in registers; X chunk staged N-major in LDS.
// ---------------------------------------------------------------------------
__global__ __launch_bounds__(128) void transform_gemm(const float* __restrict__ x,
                                                      const float* __restrict__ W,
                                                      const float* __restrict__ bias,
                                                      float* __restrict__ tr) {
    const int tid = threadIdx.x, w = tid >> 5, lane = tid & 31;
    const int n_lane = lane & 15, half = lane >> 4;
    const int p0 = blockIdx.x * 64;
    const int o0 = blockIdx.y * 64;
    const size_t base = (size_t)blockIdx.z * PB;

    __shared__ unsigned short ldsX[64 * 264];   // [p_local][k], 528B rows (16-aligned)

    // Stage X[k, p0..p0+63] -> bf16 LDS, N-major. Pairs along k, packed b32 stores.
    {
        const int pl = tid & 63;                 // lane-constant column
        const int k2b = tid >> 6;                // 0..1
        const float* gx = x + base + p0 + pl;
        unsigned short* lp = ldsX + pl * 264;
        for (int i = 0; i < 64; ++i) {
            int k2 = k2b + 2 * i;                // k = 2*k2, covers 0..255
            float a  = gx[(size_t)(2 * k2) * HW];
            float b2 = gx[(size_t)(2 * k2 + 1) * HW];
            *(unsigned int*)(lp + 2 * k2) = cvt_pk2(a, b2);
        }
    }

    // A fragments: this wave's 16 W rows, full K=256 (8 frags of 16x32).
    Frag aw[8];
    {
        const float* wrow = W + (size_t)(o0 + w * 16 + n_lane) * 256;
#pragma unroll
        for (int ks = 0; ks < 8; ++ks)
#pragma unroll
            for (int v = 0; v < 8; ++v) {
                int k = ks * 32 + kpat(v, half);
                f32x2 f = *(const f32x2*)(wrow + k);
                aw[ks].u[v] = cvt_pk2(f[0], f[1]);
            }
    }

    // Accumulators pre-loaded with bias (same value across all N columns).
    v8f acc[4];
#pragma unroll
    for (int r = 0; r < 8; ++r) {
        float bv = bias[o0 + w * 16 + r + 8 * half];
#pragma unroll
        for (int nt = 0; nt < 4; ++nt) acc[nt][r] = bv;
    }

    __syncthreads();

#pragma unroll
    for (int ks = 0; ks < 8; ++ks) {
#pragma unroll
        for (int nt = 0; nt < 4; ++nt) {
            Frag bx;
            load_frag(&ldsX[(nt * 16 + n_lane) * 264 + ks * 32], half, bx);
            acc[nt] = wmma_bf16(aw[ks].v, bx.v, acc[nt]);
        }
    }

    float* trb = tr + base;
#pragma unroll
    for (int nt = 0; nt < 4; ++nt)
#pragma unroll
        for (int r = 0; r < 8; ++r)
            trb[(size_t)(o0 + w * 16 + r + 8 * half) * HW + p0 + nt * 16 + n_lane] =
                acc[nt][r];
}

// ---------------------------------------------------------------------------
// Kernel 3: fused  Y[n,:] += relu(Xa@Xb)^2/n^4 @ TRview   (never stores S)
// Block = 4 waves, n-tile = 64 rows (16 per wave). m-loop in chunks of 32.
// Per wave: Xa A-frags (K=256, 64 VGPRs) + 16 Y accum tiles (128 VGPRs).
// ---------------------------------------------------------------------------
__global__ __launch_bounds__(128) void fused_propagate(const float* __restrict__ x,
                                                       const float* __restrict__ tr,
                                                       const float* __restrict__ inv_n4_arr,
                                                       float* __restrict__ out) {
    const int tid = threadIdx.x, w = tid >> 5, lane = tid & 31;
    const int n_lane = lane & 15, half = lane >> 4;
    const size_t base = (size_t)blockIdx.y * PB;
    const int n0 = blockIdx.x * 64 + w * 16;            // this wave's hw rows
    const float inv_n4 = inv_n4_arr[blockIdx.y];

    __shared__ unsigned short ldsXb[32 * 264];          // [m_local][k], 528B rows
    __shared__ unsigned short ldsTR[256 * 40];          // [c][m_k], 80B rows
    __shared__ unsigned short ldsS[4][16 * 40];         // per-wave S tile (16x32)

    // Persistent A fragments: Xa rows [n0, n0+16), K = 256, bf16.
    Frag af[8];
    {
        const float* xrow = x + base + (size_t)(n0 + n_lane) * 256;
#pragma unroll
        for (int ks = 0; ks < 8; ++ks)
#pragma unroll
            for (int v = 0; v < 8; ++v) {
                int k = ks * 32 + kpat(v, half);
                f32x2 f = *(const f32x2*)(xrow + k);
                af[ks].u[v] = cvt_pk2(f[0], f[1]);
            }
    }

    v8f yacc[16];
#pragma unroll
    for (int t = 0; t < 16; ++t)
#pragma unroll
        for (int r = 0; r < 8; ++r) yacc[t][r] = 0.f;

    // Per-thread constant staging coordinates.
    const int ml  = tid & 31;                 // Xb column (m_local)
    const int k2b = tid >> 5;                 // 0..3, Xb k-pair phase

    for (int mc = 0; mc < HW / 32; ++mc) {              // 72 iterations
        const int m0 = mc * 32;

        // Stage Xb chunk (K=256 x m=32): pairs along k, packed b32 stores.
        {
            const float* gx = x + base + m0 + ml;
            unsigned short* lp = ldsXb + ml * 264;
            for (int i = 0; i < 32; ++i) {
                int k2 = k2b + 4 * i;                    // k = 2*k2
                float a  = gx[(size_t)(2 * k2) * HW];
                float b2 = gx[(size_t)(2 * k2 + 1) * HW];
                *(unsigned int*)(lp + 2 * k2) = cvt_pk2(a, b2);
            }
        }
        // Stage TR chunk (m=32 rows x c=256) transposed to [c][m_k]:
        // pairs along m, packed b32 stores.
        {
            const float* gt = tr + base + (size_t)m0 * 256;
            for (int kk2 = 0; kk2 < 16; ++kk2) {
#pragma unroll
                for (int h2 = 0; h2 < 2; ++h2) {
                    int c = tid + 128 * h2;
                    float a  = gt[(size_t)(2 * kk2) * 256 + c];
                    float b2 = gt[(size_t)(2 * kk2 + 1) * 256 + c];
                    *(unsigned int*)(ldsTR + c * 40 + 2 * kk2) = cvt_pk2(a, b2);
                }
            }
        }
        // Prefetch next chunk's TR rows into cache while we compute.
        if (mc + 1 < HW / 32) {
            const float* nt_ = tr + base + (size_t)(m0 + 32) * 256;
            __builtin_prefetch(nt_ + tid * 64, 0, 1);   // 8K floats, 1 line/thread
        }
        __syncthreads();

        // cos tiles: two 16x16 tiles over K=256 (8 WMMA each).
        v8f cacc[2];
#pragma unroll
        for (int t = 0; t < 2; ++t)
#pragma unroll
            for (int r = 0; r < 8; ++r) cacc[t][r] = 0.f;

#pragma unroll
        for (int ks = 0; ks < 8; ++ks)
#pragma unroll
            for (int t = 0; t < 2; ++t) {
                Frag bx;
                load_frag(&ldsXb[(t * 16 + n_lane) * 264 + ks * 32], half, bx);
                cacc[t] = wmma_bf16(af[ks].v, bx.v, cacc[t]);
            }

        // S = relu(raw)^2 * n^-4 -> wave-private LDS (row-major 16x32),
        // converting C-layout (M striped over VGPRs) to A-fragment layout.
#pragma unroll
        for (int t = 0; t < 2; ++t)
#pragma unroll
            for (int r = 0; r < 8; ++r) {
                float f = cacc[t][r];
                f = fmaxf(f, 0.f);
                f = f * f * inv_n4;
                ldsS[w][(r + 8 * half) * 40 + t * 16 + n_lane] = cvt1(f);
            }
        // Same-wave LDS ops are in-order; read S back as 16x32 A-fragment.
        Frag sf;
        load_frag(&ldsS[w][n_lane * 40], half, sf);

        // Y += S(16x32) @ TR(32 x 256): 16 column tiles, one K=32 WMMA each.
#pragma unroll
        for (int ct = 0; ct < 16; ++ct) {
            Frag bt;
            load_frag(&ldsTR[(ct * 16 + n_lane) * 40], half, bt);
            yacc[ct] = wmma_bf16(sf.v, bt.v, yacc[ct]);
        }
        __syncthreads();
    }

    // Store Y flat at n*256 + c (raw reshape back to (c,h,w) is identity).
    float* ob = out + base;
#pragma unroll
    for (int ct = 0; ct < 16; ++ct)
#pragma unroll
        for (int r = 0; r < 8; ++r)
            ob[(size_t)(n0 + r + 8 * half) * 256 + ct * 16 + n_lane] = yacc[ct][r];
}

// ---------------------------------------------------------------------------
extern "C" void kernel_launch(void* const* d_in, const int* in_sizes, int n_in,
                              void* d_out, int out_size, void* d_ws, size_t ws_size,
                              hipStream_t stream) {
    const float* x    = (const float*)d_in[0];   // (16,256,48,48)
    const float* W    = (const float*)d_in[1];   // (256,256)
    const float* bias = (const float*)d_in[2];   // (256,)
    float* out = (float*)d_out;

    float* tr    = (float*)d_ws;                 // 16 * 256 * 2304 floats
    float* norms = tr + (size_t)16 * PB;         // 16 floats (1/n^4)

    sumsq_kernel<<<16, 256, 0, stream>>>(x, norms);

    dim3 g2(HW / 64, CC / 64, 16);               // (36, 4, 16)
    transform_gemm<<<g2, 128, 0, stream>>>(x, W, bias, tr);

    dim3 g3(HW / 64, 16);                        // (36, 16)
    fused_propagate<<<g3, 128, 0, stream>>>(x, tr, norms, out);
}
